// GINEncoder_88149908783552
// MI455X (gfx1250) — compile-verified, matrix operand order checked by reading, not run on hardware
//
#include <hip/hip_runtime.h>
#include <hip/hip_fp16.h>

#define D 128
#define LDSK 136   // padded f16 row stride (breaks LDS bank-conflict pattern on 16B reads)

typedef __attribute__((ext_vector_type(16))) _Float16 v16h;
typedef __attribute__((ext_vector_type(8)))  _Float16 v8h;
typedef __attribute__((ext_vector_type(8)))  float    v8f;

// ---------------------------------------------------------------------------
// agg = h  (initialize accumulator with self term: h = x + segment_sum(...))
// ---------------------------------------------------------------------------
__global__ void gin_copy_kernel(const float* __restrict__ src,
                                float* __restrict__ dst, long n4) {
    long i = (long)blockIdx.x * blockDim.x + threadIdx.x;
    if (i < n4) ((float4*)dst)[i] = ((const float4*)src)[i];
}

// ---------------------------------------------------------------------------
// Scatter-add: agg[dst[e]] += h[src[e]].  One wave per edge: edge id is
// wave-uniform (readfirstlane -> SGPR), each lane moves a float4 (coalesced
// 512B per edge) and issues 4 global_atomic_add_f32.  The whole 51.2MB node
// feature matrix is resident in the 192MB L2, so gather + atomics resolve
// in L2, not HBM.
// ---------------------------------------------------------------------------
__global__ void gin_scatter_kernel(const float* __restrict__ h,
                                   const int* __restrict__ srcIdx,
                                   const int* __restrict__ dstIdx,
                                   float* __restrict__ agg, int nEdges) {
    long tid  = (long)blockIdx.x * blockDim.x + threadIdx.x;
    int  edge = (int)(tid >> 5);
    if (edge >= nEdges) return;
    int f = (int)(tid & 31) * 4;

    int s = srcIdx[edge];
    int d = dstIdx[edge];
    s = __builtin_amdgcn_readfirstlane(s);   // wave-uniform -> scalar regs
    d = __builtin_amdgcn_readfirstlane(d);

    const float4 v = *(const float4*)(h + (long)s * D + f);
    float* o = agg + (long)d * D + f;
    atomicAdd(o + 0, v.x);
    atomicAdd(o + 1, v.y);
    atomicAdd(o + 2, v.z);
    atomicAdd(o + 3, v.w);
}

// ---------------------------------------------------------------------------
// Fused MLP: out = [relu]( relu(in @ W1 + b1) @ W2 + b2 )
// Block = 256 threads (8 waves), 128 rows per block, f16 WMMA, f32 accum.
// LDS: W2 tile (34KB) + union{W1 tile | activation tile} (34KB) = 68KB
//   -> 4 blocks / 320KB WGP.
// ---------------------------------------------------------------------------
__global__ __launch_bounds__(256)
void gin_mlp_kernel(const float* __restrict__ in,
                    const float* __restrict__ W1, const float* __restrict__ b1,
                    const float* __restrict__ W2, const float* __restrict__ b2,
                    float* __restrict__ out, int nRows, int finalRelu) {
    // Transposed [n][k] f16 weight tiles -> B-fragments are contiguous 16B reads.
    __shared__ _Float16 ldsW2[D * LDSK];
    __shared__ _Float16 ldsU [D * LDSK];   // W1 during GEMM1, activations after

    _Float16* ldsW1 = ldsU;
    _Float16* ldsH  = ldsU;                // aliased: W1 dead after GEMM1

    const int tid = threadIdx.x;

    // Stage weights (f32 -> f16, transposed) into LDS.
    for (int i = tid; i < D * D; i += 256) {
        int k = i >> 7, n = i & 127;
        ldsW1[n * LDSK + k] = (_Float16)W1[i];
        ldsW2[n * LDSK + k] = (_Float16)W2[i];
    }
    __syncthreads();

    const int  wid   = tid >> 5;
    const int  lane  = tid & 31;
    const int  mloc  = lane & 15;            // A: row-in-tile | C: col-in-tile
    const int  khalf = (lane >> 4) << 3;     // 0 or 8 (K offset per lane half)
    const int  radd  = (lane >> 4) << 3;     // C: +8 rows for lanes 16..31
    const long rowBase = (long)blockIdx.x * 128 + wid * 16;

    long arow = rowBase + mloc;
    if (arow >= nRows) arow = nRows - 1;     // clamp loads; stores predicated

    float bias1[8], bias2[8];
#pragma unroll
    for (int nt = 0; nt < 8; ++nt) {
        bias1[nt] = b1[nt * 16 + mloc];
        bias2[nt] = b2[nt * 16 + mloc];
    }

    v8f acc[8];
#pragma unroll
    for (int nt = 0; nt < 8; ++nt) acc[nt] = (v8f){0.f,0.f,0.f,0.f,0.f,0.f,0.f,0.f};

    // ---------------- GEMM1: in @ W1 ----------------
    const float* aptr = in + arow * D;
#pragma unroll
    for (int ks = 0; ks < 4; ++ks) {
        const int kb = ks * 32 + khalf;
        float4 f0 = *(const float4*)(aptr + kb);
        float4 f1 = *(const float4*)(aptr + kb + 4);
        float4 f2 = *(const float4*)(aptr + kb + 16);
        float4 f3 = *(const float4*)(aptr + kb + 20);
        v16h a;
        a[0]=(_Float16)f0.x; a[1]=(_Float16)f0.y; a[2]=(_Float16)f0.z; a[3]=(_Float16)f0.w;
        a[4]=(_Float16)f1.x; a[5]=(_Float16)f1.y; a[6]=(_Float16)f1.z; a[7]=(_Float16)f1.w;
        a[8]=(_Float16)f2.x; a[9]=(_Float16)f2.y; a[10]=(_Float16)f2.z; a[11]=(_Float16)f2.w;
        a[12]=(_Float16)f3.x; a[13]=(_Float16)f3.y; a[14]=(_Float16)f3.z; a[15]=(_Float16)f3.w;
#pragma unroll
        for (int nt = 0; nt < 8; ++nt) {
            const int n = nt * 16 + mloc;
            const _Float16* wp = &ldsW1[n * LDSK + ks * 32 + khalf];
            v8h blo = *(const v8h*)(wp);
            v8h bhi = *(const v8h*)(wp + 16);
            v16h b = __builtin_shufflevector(blo, bhi,
                      0,1,2,3,4,5,6,7,8,9,10,11,12,13,14,15);
            acc[nt] = __builtin_amdgcn_wmma_f32_16x16x32_f16(
                          false, a, false, b, (short)0, acc[nt], false, false);
        }
    }

    // All waves done reading W1 -> safe to overwrite aliased region with
    // activations.
    __syncthreads();

    // bias + relu, stage f16 activation rows in LDS (each wave owns its rows)
#pragma unroll
    for (int nt = 0; nt < 8; ++nt) {
        const int col = nt * 16 + mloc;
#pragma unroll
        for (int r = 0; r < 8; ++r) {
            float v = acc[nt][r] + bias1[nt];
            v = v > 0.f ? v : 0.f;
            ldsH[(wid * 16 + r + radd) * LDSK + col] = (_Float16)v;
        }
    }

    // ---------------- GEMM2: ldsH @ W2 ----------------
#pragma unroll
    for (int nt = 0; nt < 8; ++nt) acc[nt] = (v8f){0.f,0.f,0.f,0.f,0.f,0.f,0.f,0.f};

    const _Float16* hrow = &ldsH[(wid * 16 + mloc) * LDSK];
#pragma unroll
    for (int ks = 0; ks < 4; ++ks) {
        const int kb = ks * 32 + khalf;
        v8h alo = *(const v8h*)(hrow + kb);
        v8h ahi = *(const v8h*)(hrow + kb + 16);
        v16h a = __builtin_shufflevector(alo, ahi,
                  0,1,2,3,4,5,6,7,8,9,10,11,12,13,14,15);
#pragma unroll
        for (int nt = 0; nt < 8; ++nt) {
            const int n = nt * 16 + mloc;
            const _Float16* wp = &ldsW2[n * LDSK + ks * 32 + khalf];
            v8h blo = *(const v8h*)(wp);
            v8h bhi = *(const v8h*)(wp + 16);
            v16h b = __builtin_shufflevector(blo, bhi,
                      0,1,2,3,4,5,6,7,8,9,10,11,12,13,14,15);
            acc[nt] = __builtin_amdgcn_wmma_f32_16x16x32_f16(
                          false, a, false, b, (short)0, acc[nt], false, false);
        }
    }

    // bias + optional relu, store f32 output
#pragma unroll
    for (int nt = 0; nt < 8; ++nt) {
        const int col = nt * 16 + mloc;
#pragma unroll
        for (int r = 0; r < 8; ++r) {
            float v = acc[nt][r] + bias2[nt];
            if (finalRelu) v = v > 0.f ? v : 0.f;
            long row = rowBase + r + radd;
            if (row < nRows) out[row * D + col] = v;
        }
    }
}

// ---------------------------------------------------------------------------
extern "C" void kernel_launch(void* const* d_in, const int* in_sizes, int n_in,
                              void* d_out, int out_size, void* d_ws, size_t ws_size,
                              hipStream_t stream) {
    const int N = in_sizes[0] / D;       // 100000
    const int E = in_sizes[1] / 2;       // 1600000

    const float* x   = (const float*)d_in[0];
    const int*   ei  = (const int*)d_in[1];
    const int*   src = ei;               // edge_index[0]
    const int*   dst = ei + E;           // edge_index[1]

    float* ws  = (float*)d_ws;           // N*D f32 accumulator buffer
    float* out = (float*)d_out;          // N*D f32, doubles as activation ping

    const long n4 = (long)N * D / 4;
    dim3 blk(256);
    dim3 gCopy((unsigned)((n4 + 255) / 256));
    dim3 gScat((unsigned)(((long)E * 32 + 255) / 256));
    dim3 gMlp((unsigned)((N + 127) / 128));

    const float* hin = x;
    for (int l = 0; l < 3; ++l) {
        const float* W1 = (const float*)d_in[2 + 4 * l + 0];
        const float* b1 = (const float*)d_in[2 + 4 * l + 1];
        const float* W2 = (const float*)d_in[2 + 4 * l + 2];
        const float* b2 = (const float*)d_in[2 + 4 * l + 3];

        // agg = h ;  agg[dst] += h[src] ;  out = MLP(agg)
        gin_copy_kernel   <<<gCopy, blk, 0, stream>>>(hin, ws, n4);
        gin_scatter_kernel<<<gScat, blk, 0, stream>>>(hin, src, dst, ws, E);
        gin_mlp_kernel    <<<gMlp,  blk, 0, stream>>>(ws, W1, b1, W2, b2,
                                                      out, N, l < 2 ? 1 : 0);
        hin = out;
    }
}
